// GCN_13975823581721
// MI455X (gfx1250) — matrix operand order repaired and checked by reading, not compiled, not run on hardware
//
#include <hip/hip_runtime.h>
#include <hip/hip_bf16.h>

#define NNODES 50000
#define DFEAT  128
#define SLOPE  0.01f

typedef __attribute__((ext_vector_type(16))) __bf16 v16bf;
typedef __attribute__((ext_vector_type(8)))  float  v8f;

// ---------------------------------------------------------------- utilities
__global__ void zero_f32(float* __restrict__ p, int n) {
    int i = blockIdx.x * blockDim.x + threadIdx.x;
    if (i < n) p[i] = 0.0f;
}

__global__ void count_deg(const int* __restrict__ src, const int* __restrict__ dst,
                          float* __restrict__ outdeg, float* __restrict__ indeg, int e) {
    int i = blockIdx.x * blockDim.x + threadIdx.x;
    if (i < e) {
        atomicAdd(&outdeg[src[i]], 1.0f);
        atomicAdd(&indeg[dst[i]], 1.0f);
    }
}

__global__ void finish_deg(float* __restrict__ outdeg, float* __restrict__ indeg, int n) {
    int i = blockIdx.x * blockDim.x + threadIdx.x;
    if (i < n) {
        outdeg[i] = rsqrtf(fmaxf(outdeg[i], 1.0f));
        indeg[i]  = rsqrtf(fmaxf(indeg[i], 1.0f));
    }
}

// ------------------------------------------------- weight pre-pack (once per layer)
// Converts W (128x128 f32, row-major) to bf16 laid out exactly as WMMA B fragments:
// fragment index t = tile*128 + kq*32 + lane holds 16 bf16 (32B) for
// column nc = tile*16 + lane%16, K = kq*32 + (lane>=16?8:0) + e + (e>=8?8:0).
__global__ void pack_w(const float* __restrict__ W, __bf16* __restrict__ Wp) {
    int t = blockIdx.x * blockDim.x + threadIdx.x;
    if (t >= 1024) return;
    int lane  = t & 31;
    int kq    = (t >> 5) & 3;
    int tile  = t >> 7;
    int laneM = lane & 15;
    int hi    = lane >> 4;
    int nc    = tile * 16 + laneM;
    int kb    = kq * 32 + hi * 8;
    __bf16* dst = Wp + (size_t)t * 16;
#pragma unroll
    for (int e = 0; e < 16; ++e)
        dst[e] = (__bf16)W[(size_t)(kb + e + (e >= 8 ? 8 : 0)) * DFEAT + nc];
}

// ------------------------------------------------- WMMA GEMM, one 16x16 tile / wave
// out[r][c] = ( pre(A)[r][:] @ W[:][c] ) * isq_out[r]
// pre(A)    = PRE ? LeakyReLU(A*isq_in[r] + bias[k]) : A   (fused prev-layer epilogue)
template <bool PRE>
__global__ __launch_bounds__(256) void gcn_gemm(
    const float* __restrict__ A, const __bf16* __restrict__ Wp,
    const float* __restrict__ isq_out, const float* __restrict__ isq_in,
    const float* __restrict__ bias, float* __restrict__ out)
{
    const int wave  = threadIdx.x >> 5;
    const int lane  = threadIdx.x & 31;
    const int laneM = lane & 15;
    const int hi    = lane >> 4;               // 0: lanes 0-15, 1: lanes 16-31
    const int row   = blockIdx.x * 16 + laneM; // A row this lane feeds
    const int col0  = wave * 16;               // this wave's output column tile

    const float* __restrict__ arow = A + (size_t)row * DFEAT;
    const float sIn = PRE ? isq_in[row] : 1.0f;
    const v16bf* __restrict__ wfrag = (const v16bf*)Wp + (size_t)wave * 128 + lane;

    v8f acc = {};
#pragma unroll
    for (int kq = 0; kq < 4; ++kq) {
        const int ka = kq * 32 + hi * 8;
        // A fragment: two groups of 8 consecutive f32 -> 4 x b128 loads
        float4 a0 = *(const float4*)(arow + ka);
        float4 a1 = *(const float4*)(arow + ka + 4);
        float4 a2 = *(const float4*)(arow + ka + 16);
        float4 a3 = *(const float4*)(arow + ka + 20);
        float av[16] = {a0.x, a0.y, a0.z, a0.w, a1.x, a1.y, a1.z, a1.w,
                        a2.x, a2.y, a2.z, a2.w, a3.x, a3.y, a3.z, a3.w};
        if (PRE) {
            float4 b0 = *(const float4*)(bias + ka);
            float4 b1 = *(const float4*)(bias + ka + 4);
            float4 b2 = *(const float4*)(bias + ka + 16);
            float4 b3 = *(const float4*)(bias + ka + 20);
            float bv[16] = {b0.x, b0.y, b0.z, b0.w, b1.x, b1.y, b1.z, b1.w,
                            b2.x, b2.y, b2.z, b2.w, b3.x, b3.y, b3.z, b3.w};
#pragma unroll
            for (int e = 0; e < 16; ++e) {
                float v = av[e] * sIn + bv[e];
                av[e] = (v >= 0.0f) ? v : SLOPE * v;
            }
        }
        v16bf afrag;
#pragma unroll
        for (int e = 0; e < 16; ++e) afrag[e] = (__bf16)av[e];

        v16bf bfrag = wfrag[kq * 32];   // contiguous 32B pre-packed fragment

        acc = __builtin_amdgcn_wmma_f32_16x16x32_bf16(
            /*neg_a=*/false, afrag, /*neg_b=*/false, bfrag,
            /*c_mod=*/(short)0, acc, /*reuse_a=*/false, /*reuse_b=*/false);
    }

    // store C/D: VGPR g -> row (g + hi*8), col laneM; scale by isq_out[row]
    const int rbase = blockIdx.x * 16 + hi * 8;
#pragma unroll
    for (int g = 0; g < 8; ++g) {
        const int rr = rbase + g;
        out[(size_t)rr * DFEAT + col0 + laneM] = acc[g] * isq_out[rr];
    }
}

// ------------------------------------------------- edge scatter: agg[dst] += hs[src]
// 32 lanes per edge, float4 per lane -> 4 f32 atomics; L2-resident working set.
__global__ __launch_bounds__(256) void scatter_edges(
    const float* __restrict__ hs, const int* __restrict__ src,
    const int* __restrict__ dst, float* __restrict__ agg, int e)
{
    int t  = blockIdx.x * blockDim.x + threadIdx.x;
    int ei = t >> 5;
    if (ei < e) {
        int q = (t & 31) << 2;
        int s = src[ei], d = dst[ei];
        const float4 v = *reinterpret_cast<const float4*>(hs + (size_t)s * DFEAT + q);
        float* o = agg + (size_t)d * DFEAT + q;
        atomicAdd(o + 0, v.x);
        atomicAdd(o + 1, v.y);
        atomicAdd(o + 2, v.z);
        atomicAdd(o + 3, v.w);
    }
}

// ------------------------------------------------- head: LeakyReLU epilogue + 128->4
__global__ void final_linear(const float* __restrict__ agg, const float* __restrict__ isq_in,
                             const float* __restrict__ b5, const float* __restrict__ Wl,
                             const float* __restrict__ bl, float* __restrict__ out, int n)
{
    int t = blockIdx.x * blockDim.x + threadIdx.x;
    if (t < n * 4) {
        int r = t >> 2, c = t & 3;
        float s = isq_in[r];
        const float* row = agg + (size_t)r * DFEAT;
        float acc = bl[c];
#pragma unroll 4
        for (int k = 0; k < DFEAT; ++k) {
            float v = row[k] * s + b5[k];
            v = (v >= 0.0f) ? v : SLOPE * v;
            acc = fmaf(v, Wl[k * 4 + c], acc);
        }
        out[t] = acc;
    }
}

// ---------------------------------------------------------------- driver
extern "C" void kernel_launch(void* const* d_in, const int* in_sizes, int n_in,
                              void* d_out, int out_size, void* d_ws, size_t ws_size,
                              hipStream_t stream) {
    const float* x   = (const float*)d_in[0];
    const int*   src = (const int*)d_in[1];
    const int*   dst = (const int*)d_in[2];
    const float* W[5] = {(const float*)d_in[3], (const float*)d_in[5], (const float*)d_in[7],
                         (const float*)d_in[9], (const float*)d_in[11]};
    const float* b[5] = {(const float*)d_in[4], (const float*)d_in[6], (const float*)d_in[8],
                         (const float*)d_in[10], (const float*)d_in[12]};
    const float* Wl = (const float*)d_in[13];
    const float* bl = (const float*)d_in[14];
    float* out = (float*)d_out;

    const int E = in_sizes[1];

    float* ws      = (float*)d_ws;
    float* isq_out = ws;                                   // N
    float* isq_in  = ws + NNODES;                          // N
    float* hs      = ws + 2 * (size_t)NNODES;              // N*128
    float* agg     = hs + (size_t)NNODES * DFEAT;          // N*128
    __bf16* Wp     = (__bf16*)(agg + (size_t)NNODES * DFEAT); // 128*128 bf16 (32KB)

    // degrees -> inv-sqrt (clamped to 1)
    zero_f32<<<(2 * NNODES + 255) / 256, 256, 0, stream>>>(isq_out, 2 * NNODES);
    count_deg<<<(E + 255) / 256, 256, 0, stream>>>(src, dst, isq_out, isq_in, E);
    finish_deg<<<(NNODES + 255) / 256, 256, 0, stream>>>(isq_out, isq_in, NNODES);

    const int gemm_grid = NNODES / 16;            // 50000 % 16 == 0
    const int nfeat     = NNODES * DFEAT;
    const int scat_grid = (E * 32 + 255) / 256;

    const float* cur = x;
    for (int l = 0; l < 5; ++l) {
        pack_w<<<4, 256, 0, stream>>>(W[l], Wp);
        if (l == 0)
            gcn_gemm<false><<<gemm_grid, 256, 0, stream>>>(
                cur, Wp, isq_out, isq_in, nullptr, hs);
        else
            gcn_gemm<true><<<gemm_grid, 256, 0, stream>>>(
                cur, Wp, isq_out, isq_in, b[l - 1], hs);
        zero_f32<<<(nfeat + 255) / 256, 256, 0, stream>>>(agg, nfeat);
        scatter_edges<<<scat_grid, 256, 0, stream>>>(hs, src, dst, agg, E);
        cur = agg;
    }
    final_linear<<<(NNODES * 4 + 255) / 256, 256, 0, stream>>>(
        agg, isq_in, b[4], Wl, bl, out, NNODES);
}
